// GNNCell_74947179316229
// MI455X (gfx1250) — matrix operand order repaired and checked by reading, not compiled, not run on hardware
//
#include <hip/hip_runtime.h>

#define N_NODES 50000
#define N_EDGES 800000
#define DIM     128
#define LEAKY   0.2f
#define BN_EPS  1e-5f

typedef __attribute__((ext_vector_type(16))) _Float16 v16h;
typedef __attribute__((ext_vector_type(8)))  float    v8f;

// ---------------------------------------------------------------- utilities
__global__ void k_zero_u32(unsigned* __restrict__ p, int n) {
    int i = blockIdx.x * blockDim.x + threadIdx.x;
    if (i < n) p[i] = 0u;
}

// ------------------------------------------------------- degree histograms
__global__ void k_degrees(const int* __restrict__ src, const int* __restrict__ dst,
                          unsigned* __restrict__ degO, unsigned* __restrict__ degI) {
    int e = blockIdx.x * blockDim.x + threadIdx.x;
    if (e < N_EDGES) {
        atomicAdd(&degO[src[e]], 1u);
        atomicAdd(&degI[dst[e]], 1u);
    }
}

// ------------------------------------------- exclusive scan of in-degrees
// Single block, chunked Hillis-Steele; writes row_ptr[N+1] and cursor[N].
__global__ void k_scan(const unsigned* __restrict__ degI,
                       unsigned* __restrict__ row_ptr,
                       unsigned* __restrict__ cursor) {
    __shared__ unsigned s[256];
    const int t = threadIdx.x;
    unsigned running = 0;
    for (int base = 0; base < N_NODES; base += 256) {
        unsigned x = (base + t < N_NODES) ? degI[base + t] : 0u;
        s[t] = x;
        __syncthreads();
        for (int off = 1; off < 256; off <<= 1) {
            unsigned v = (t >= off) ? s[t - off] : 0u;
            __syncthreads();
            s[t] += v;
            __syncthreads();
        }
        unsigned incl = s[t];
        unsigned tot  = s[255];
        if (base + t < N_NODES) {
            unsigned ex = running + incl - x;
            row_ptr[base + t] = ex;
            cursor[base + t]  = ex;
        }
        running += tot;
        __syncthreads();
    }
    if (t == 0) row_ptr[N_NODES] = running;
}

// --------------------------------------------------- rsqrt(max(deg,1)) LUTs
__global__ void k_invdeg(const unsigned* __restrict__ degO, const unsigned* __restrict__ degI,
                         float* __restrict__ invO, float* __restrict__ invI) {
    int i = blockIdx.x * blockDim.x + threadIdx.x;
    if (i < N_NODES) {
        unsigned a = degO[i]; if (a == 0u) a = 1u;
        unsigned b = degI[i]; if (b == 0u) b = 1u;
        invO[i] = rsqrtf((float)a);
        invI[i] = rsqrtf((float)b);
    }
}

// ------------------------------------------------------------- CSR filling
__global__ void k_fill(const int* __restrict__ src, const int* __restrict__ dst,
                       unsigned* __restrict__ cursor, int* __restrict__ csrSrc) {
    int e = blockIdx.x * blockDim.x + threadIdx.x;
    if (e < N_EDGES) {
        unsigned pos = atomicAdd(&cursor[dst[e]], 1u);
        csrSrc[pos] = src[e];
    }
}

// ------------------------------------------- per-destination gather (no atomics)
// One 128-thread block per node; thread = feature. Coalesced 512B row reads,
// edge ids + source scales staged through LDS.
__global__ __launch_bounds__(128)
void k_gather(const float* __restrict__ V, const int* __restrict__ csrSrc,
              const unsigned* __restrict__ row_ptr,
              const float* __restrict__ invO, const float* __restrict__ invI,
              float* __restrict__ agg) {
    __shared__ int   sE[128];
    __shared__ float sS[128];
    const int f   = threadIdx.x;
    const int nid = blockIdx.x;
    const unsigned s0 = row_ptr[nid];
    const unsigned s1 = row_ptr[nid + 1];
    float acc = 0.f;
    for (unsigned base = s0; base < s1; base += 128u) {
        unsigned rem = s1 - base;
        int cnt = (rem < 128u) ? (int)rem : 128;
        __syncthreads();
        if (f < cnt) {
            int sid = csrSrc[base + f];
            sE[f] = sid;
            sS[f] = invO[sid];
        }
        __syncthreads();
        for (int j = 0; j < cnt; ++j)
            acc += V[(size_t)sE[j] * DIM + f] * sS[j];
    }
    agg[(size_t)nid * DIM + f] = acc * invI[nid];
}

// ------------------------- z = agg @ W + b  via split-f16 WMMA, + BN partials
// One wave per 16x16 tile; 8 waves/block -> 16 rows x 128 cols per block.
// Split trick: x = hi + lo (f16 each); x*y ~ hi*hi + hi*lo + lo*hi, f32 accum.
__global__ __launch_bounds__(256)
void k_gemm_bnstats(const float* __restrict__ agg, const float* __restrict__ W,
                    const float* __restrict__ bias, float* __restrict__ z,
                    float* __restrict__ bnSum, float* __restrict__ bnSq) {
    const int lane = threadIdx.x & 31;
    const int wave = threadIdx.x >> 5;   // column tile 0..7
    const int half = lane >> 4;          // lane half
    const int ln   = lane & 15;
    const int row0 = blockIdx.x * 16;
    const int n0   = wave * 16;

    v8f c = {};
    for (int kb = 0; kb < DIM; kb += 32) {
        v16h aHi, aLo, bHi, bLo;
        // A fragment: 16x32 f16 layout (ISA 7.12.2): lane ln = row M,
        // vgpr v<4 -> K = 8*half + 2v ; v>=4 -> K = 16 + 8*half + 2(v-4)
        #pragma unroll
        for (int v = 0; v < 8; ++v) {
            int k = kb + ((v < 4) ? (8 * half + 2 * v) : (16 + 8 * half + 2 * (v - 4)));
            const float* ap = agg + (size_t)(row0 + ln) * DIM + k;
            float a0 = ap[0], a1 = ap[1];
            _Float16 h0 = (_Float16)a0, h1 = (_Float16)a1;
            aHi[2 * v]     = h0;
            aHi[2 * v + 1] = h1;
            aLo[2 * v]     = (_Float16)(a0 - (float)h0);
            aLo[2 * v + 1] = (_Float16)(a1 - (float)h1);
        }
        // B fragment: 32x16, lane = column N; same K order as A:
        // vgpr v<4 -> K = 8*half + 2v ; v>=4 -> K = 16 + 8*half + 2(v-4)
        #pragma unroll
        for (int v = 0; v < 8; ++v) {
            int k = kb + ((v < 4) ? (8 * half + 2 * v) : (16 + 8 * half + 2 * (v - 4)));
            float b0 = W[(size_t)k * DIM + n0 + ln];
            float b1 = W[(size_t)(k + 1) * DIM + n0 + ln];
            _Float16 h0 = (_Float16)b0, h1 = (_Float16)b1;
            bHi[2 * v]     = h0;
            bHi[2 * v + 1] = h1;
            bLo[2 * v]     = (_Float16)(b0 - (float)h0);
            bLo[2 * v + 1] = (_Float16)(b1 - (float)h1);
        }
        c = __builtin_amdgcn_wmma_f32_16x16x32_f16(false, aHi, false, bHi, (short)0, c, false, false);
        c = __builtin_amdgcn_wmma_f32_16x16x32_f16(false, aHi, false, bLo, (short)0, c, false, false);
        c = __builtin_amdgcn_wmma_f32_16x16x32_f16(false, aLo, false, bHi, (short)0, c, false, false);
    }

    const float bv = bias[n0 + ln];
    float sum = 0.f, sq = 0.f;
    // C/D layout: vgpr i -> M = i + 8*half, N = ln
    #pragma unroll
    for (int i = 0; i < 8; ++i) {
        float val = c[i] + bv;
        z[(size_t)(row0 + 8 * half + i) * DIM + n0 + ln] = val;
        sum += val;
        sq  += val * val;
    }
    atomicAdd(&bnSum[n0 + ln], sum);
    atomicAdd(&bnSq[n0 + ln],  sq);
}

// ---------------------- BN (batch stats) + LeakyReLU + residual, in place on z
__global__ void k_final(const float* __restrict__ V, const float* __restrict__ gamma,
                        const float* __restrict__ beta, const float* __restrict__ bnSum,
                        const float* __restrict__ bnSq, float* __restrict__ out) {
    int tid = blockIdx.x * blockDim.x + threadIdx.x;     // over N*D/4
    if (tid >= (N_NODES * DIM) / 4) return;
    const int base = tid * 4;
    const int f4   = (base % DIM) / 4;
    const float invN = 1.0f / (float)N_NODES;

    float4 s  = ((const float4*)bnSum)[f4];
    float4 q  = ((const float4*)bnSq)[f4];
    float4 g  = ((const float4*)gamma)[f4];
    float4 be = ((const float4*)beta)[f4];
    float4 zv = ((const float4*)out)[tid];
    float4 vv = ((const float4*)V)[tid];

    float4 o;
    {
        float m = s.x * invN; float var = fmaxf(q.x * invN - m * m, 0.f);
        float t = g.x * ((zv.x - m) * rsqrtf(var + BN_EPS)) + be.x;
        o.x = (t >= 0.f ? t : LEAKY * t) + vv.x;
    }
    {
        float m = s.y * invN; float var = fmaxf(q.y * invN - m * m, 0.f);
        float t = g.y * ((zv.y - m) * rsqrtf(var + BN_EPS)) + be.y;
        o.y = (t >= 0.f ? t : LEAKY * t) + vv.y;
    }
    {
        float m = s.z * invN; float var = fmaxf(q.z * invN - m * m, 0.f);
        float t = g.z * ((zv.z - m) * rsqrtf(var + BN_EPS)) + be.z;
        o.z = (t >= 0.f ? t : LEAKY * t) + vv.z;
    }
    {
        float m = s.w * invN; float var = fmaxf(q.w * invN - m * m, 0.f);
        float t = g.w * ((zv.w - m) * rsqrtf(var + BN_EPS)) + be.w;
        o.w = (t >= 0.f ? t : LEAKY * t) + vv.w;
    }
    ((float4*)out)[tid] = o;
}

// --------------------------------------------------------------------------
extern "C" void kernel_launch(void* const* d_in, const int* in_sizes, int n_in,
                              void* d_out, int out_size, void* d_ws, size_t ws_size,
                              hipStream_t stream) {
    (void)in_sizes; (void)n_in; (void)out_size; (void)ws_size;

    const float* V     = (const float*)d_in[0];
    const float* W     = (const float*)d_in[1];
    const float* bias  = (const float*)d_in[2];
    const float* gamma = (const float*)d_in[3];
    const float* beta  = (const float*)d_in[4];
    const int*   src   = (const int*)d_in[5];
    const int*   dst   = (const int*)d_in[6];
    float* out = (float*)d_out;

    // ---- workspace carve-up (512B aligned), ~30 MB total; z lives in d_out
    char* ws = (char*)d_ws;
    size_t off = 0;
    auto carve = [&](size_t bytes) -> char* {
        char* p = ws + off;
        off = (off + bytes + 511) & ~(size_t)511;
        return p;
    };
    unsigned* deg     = (unsigned*)carve(sizeof(unsigned) * 2 * N_NODES);
    unsigned* degO    = deg;
    unsigned* degI    = deg + N_NODES;
    unsigned* row_ptr = (unsigned*)carve(sizeof(unsigned) * (N_NODES + 1));
    unsigned* cursor  = (unsigned*)carve(sizeof(unsigned) * N_NODES);
    int*      csrSrc  = (int*)carve(sizeof(int) * N_EDGES);
    float*    invO    = (float*)carve(sizeof(float) * N_NODES);
    float*    invI    = (float*)carve(sizeof(float) * N_NODES);
    float*    bn      = (float*)carve(sizeof(float) * 2 * DIM);
    float*    bnSum   = bn;
    float*    bnSq    = bn + DIM;
    float*    agg     = (float*)carve(sizeof(float) * (size_t)N_NODES * DIM);

    // 1. zero accumulators
    k_zero_u32<<<(2 * N_NODES + 255) / 256, 256, 0, stream>>>(deg, 2 * N_NODES);
    k_zero_u32<<<1, 256, 0, stream>>>((unsigned*)bn, 2 * DIM);
    // 2. degree histograms
    k_degrees<<<(N_EDGES + 255) / 256, 256, 0, stream>>>(src, dst, degO, degI);
    // 3. exclusive scan of in-degrees -> row_ptr, cursor
    k_scan<<<1, 256, 0, stream>>>(degI, row_ptr, cursor);
    // 4. rsqrt(max(deg,1))
    k_invdeg<<<(N_NODES + 255) / 256, 256, 0, stream>>>(degO, degI, invO, invI);
    // 5. CSR fill (sort edges by destination)
    k_fill<<<(N_EDGES + 255) / 256, 256, 0, stream>>>(src, dst, cursor, csrSrc);
    // 6. gather-aggregate (no feature atomics)
    k_gather<<<N_NODES, 128, 0, stream>>>(V, csrSrc, row_ptr, invO, invI, agg);
    // 7. GEMM (split-f16 WMMA, f32 accum) + bias + BN partial sums; z -> d_out
    k_gemm_bnstats<<<N_NODES / 16, 256, 0, stream>>>(agg, W, bias, out, bnSum, bnSq);
    // 8. BN + LeakyReLU + residual (in place on d_out)
    k_final<<<((N_NODES * DIM) / 4 + 255) / 256, 256, 0, stream>>>(V, gamma, beta, bnSum, bnSq, out);
}
